// GumbelSSTwithNews_523986010215
// MI455X (gfx1250) — matrix operand order repaired and checked by reading, not compiled, not run on hardware
//
#include <hip/hip_runtime.h>

// ---------------------------------------------------------------------------
// MI455X (gfx1250, wave32) BiLSTM + Gumbel head.
// 167 GFLOP of GEMM, ~25 MB working set (L2-resident) -> compute-bound on the
// matrix pipes. f16 WMMA (f32 accumulate). Recurrent h-state double-buffered
// in LDS (f16, read panel / write panel swap each step -> single barrier per
// step); cell state c in WMMA C-layout VGPRs; per-step x panel double-
// buffered in LDS and filled by the Tensor Data Mover (s_wait_tensorcnt).
// ---------------------------------------------------------------------------

typedef __attribute__((ext_vector_type(16))) _Float16 v16h;
typedef __attribute__((ext_vector_type(8)))  _Float16 v8h;
typedef __attribute__((ext_vector_type(2)))  _Float16 h2;
typedef __attribute__((ext_vector_type(8)))  float    v8f;
typedef __attribute__((ext_vector_type(4)))  unsigned int u32x4;
typedef __attribute__((ext_vector_type(8)))  int      i32x8;
typedef __attribute__((ext_vector_type(4)))  int      i32x4;

#define S_STEPS 128
#define B_ROWS  64
#define E_DIM   300
#define KX      320          // E padded to multiple of 32
#define KH      512          // H
#define KTOT    (KX + KH)    // 832
#define NGATE   2048         // 4*H
#define GCOLS   3072         // 6*H
#define XPANEL  (B_ROWS * KX)   // 20480 f16 elements = 40 KB
#define HPANEL  (B_ROWS * KH)   // 32768 f16 elements = 64 KB

// ---- WMMA fragment loaders (layouts per CDNA5 ISA 7.12.2) ------

// A fragment (16x32 f16): lane L -> row m = L&15, sel = L>>4.
// halves[0..7]  = A[m][k0 + sel*8 + 0..7]
// halves[8..15] = A[m][k0 + 16 + sel*8 + 0..7]
__device__ __forceinline__ v16h load_afrag(const _Float16* rowk) {
  union { v16h v; v8h h[2]; } u;
  u.h[0] = *(const v8h*)(rowk);
  u.h[1] = *(const v8h*)(rowk + 16);
  return u.v;
}

// B fragment (32x16 f16): lane L -> col n0+(L&15), sel = L>>4;
// halves[0..7] = K rows k0+sel*8+0..7, halves[8..15] = k0+16+sel*8+0..7.
__device__ __forceinline__ v16h load_bfrag(const _Float16* __restrict__ Bm,
                                           int k, int n0, int lane) {
  const _Float16* p = Bm + (size_t)(k + (lane >> 4) * 8) * NGATE + n0 + (lane & 15);
  v16h v;
  #pragma unroll
  for (int s = 0; s < 8; ++s) {
    v[s] = p[(size_t)s * NGATE];
    v[8 + s] = p[(size_t)(16 + s) * NGATE];
  }
  return v;
}

// One gate column-tile (16 cols at n0) for all 4 M-tiles (M=64).
// Both A panels (x and h) live in LDS -> pure 32-bit ds addressing.
__device__ __forceinline__ void gate_gemm(v8f acc[4], int n0, float bn,
    const _Float16* xp,                   // LDS x panel, stride KX (f16)
    const _Float16* hr,                   // LDS h read panel, stride KH (f16)
    const _Float16* __restrict__ Bm,      // global K-major weights (KTOT x NGATE)
    int lane)
{
  const int m = lane & 15, sel = lane >> 4;
  v8f binit;
  #pragma unroll
  for (int i = 0; i < 8; ++i) binit[i] = bn;
  #pragma unroll
  for (int mt = 0; mt < 4; ++mt) acc[mt] = binit;

  // x-projection: K tiles 0..9 (x panel staged in LDS by the TDM)
  #pragma unroll 1
  for (int kt = 0; kt < KX / 32; ++kt) {
    const int k0 = kt * 32;
    v16h bf = load_bfrag(Bm, k0, n0, lane);
    #pragma unroll
    for (int mt = 0; mt < 4; ++mt) {
      v16h af = load_afrag(xp + (mt * 16 + m) * KX + k0 + sel * 8);
      acc[mt] = __builtin_amdgcn_wmma_f32_16x16x32_f16(
          false, af, false, bf, (short)0, acc[mt], false, false);
    }
  }
  // h-projection: K tiles from LDS h read panel
  #pragma unroll 1
  for (int kt = 0; kt < KH / 32; ++kt) {
    const int k0 = kt * 32;
    v16h bf = load_bfrag(Bm, KX + k0, n0, lane);
    #pragma unroll
    for (int mt = 0; mt < 4; ++mt) {
      v16h af = load_afrag(hr + (mt * 16 + m) * KH + k0 + sel * 8);
      acc[mt] = __builtin_amdgcn_wmma_f32_16x16x32_f16(
          false, af, false, bf, (short)0, acc[mt], false, false);
    }
  }
}

__device__ __forceinline__ float sigm(float x) {
  return 1.0f / (1.0f + __expf(-x));
}

// ---- Tensor Data Mover: 1-D contiguous copy global -> LDS -----------------
// D# packing per CDNA5 ISA (groups 0/1; 2D+
// groups zero). tile_dim0 = tensor_dim0 = nelem (< 65536), data_size = 2B.
// 6-arg builtin form (clang-23 / therock-10.0 headers).
__device__ __forceinline__ void tdm_load_1d(const void* gsrc, unsigned lds_off,
                                            unsigned nelem) {
  const unsigned long long ga = (unsigned long long)(uintptr_t)gsrc;
  u32x4 g0 = {0u, 0u, 0u, 0u};
  g0[0] = 1u;                                            // count=1 (valid D#)
  g0[1] = lds_off;                                       // D#.lds_addr
  g0[2] = (unsigned)ga;                                  // global_addr[31:0]
  g0[3] = (unsigned)((ga >> 32) & 0x01FFFFFFu) | (2u << 30); // addr[56:32]|type=2
  i32x8 g1 = {0, 0, 0, 0, 0, 0, 0, 0};
  g1[0] = 1 << 16;                                       // data_size=1 -> 2B
  g1[1] = (int)((nelem & 0xFFFFu) << 16);                // tensor_dim0[15:0]
  g1[2] = (int)(nelem >> 16);                            // tensor_dim0[31:16]
  g1[3] = (int)(nelem << 16);                            // tile_dim0
  g1[5] = (int)nelem;                                    // dim0 stride (unused)
  i32x4 z4 = {0, 0, 0, 0};
  i32x8 z8 = {0, 0, 0, 0, 0, 0, 0, 0};
  __builtin_amdgcn_tensor_load_to_lds(g0, g1, z4, z4, z8, 0);
}

// ---- Kernel 1: embedding gather + f32 -> f16 (pad 300 -> 320) -------------
__global__ void k_embed(const int* __restrict__ sent,
                        const float* __restrict__ emb,
                        _Float16* __restrict__ X) {
  const int rb = blockIdx.x;                  // s*B + b, 0..8191
  const int tok = sent[rb];
  const float* e = emb + (size_t)tok * E_DIM;
  _Float16* xr = X + (size_t)rb * KX;
  for (int k = threadIdx.x; k < KX; k += blockDim.x)
    xr[k] = (k < E_DIM) ? (_Float16)e[k] : (_Float16)0.f;
}

// ---- Kernel 2: build K-major f16 weight panel [Wih ; 0 ; Whh] -------------
__global__ void k_wconv(const float* __restrict__ Wih,
                        const float* __restrict__ Whh,
                        _Float16* __restrict__ Bm) {
  const int k = blockIdx.x;                   // 0..KTOT-1
  _Float16* dst = Bm + (size_t)k * NGATE;
  if (k < E_DIM) {
    for (int n = threadIdx.x; n < NGATE; n += blockDim.x)
      dst[n] = (_Float16)Wih[(size_t)n * E_DIM + k];
  } else if (k < KX) {
    for (int n = threadIdx.x; n < NGATE; n += blockDim.x)
      dst[n] = (_Float16)0.f;
  } else {
    const int kk = k - KX;
    for (int n = threadIdx.x; n < NGATE; n += blockDim.x)
      dst[n] = (_Float16)Whh[(size_t)n * KH + kk];
  }
}

// ---- Kernel 3: one LSTM direction per 1024-thread workgroup ---------------
// Wave w owns hidden slice j in [16w,16w+16) across all four gates, so the
// cell update is wave-local. h is double-buffered in LDS: step t reads
// hpan[t&1] and writes hpan[(t+1)&1], so only ONE barrier per step is needed
// (write->read ordering at the loop top). The per-step x panel is double-
// buffered in LDS and filled by the TDM while the current step computes.
__global__ __launch_bounds__(1024, 1) void k_bilstm(
    const _Float16* __restrict__ X,        // (S*B, KX) f16
    const _Float16* __restrict__ Bmats,    // (6, KTOT, NGATE) f16
    const float* b0, const float* b1, const float* b2,
    const float* b3, const float* b4, const float* b5,
    float* __restrict__ g_inp)             // (B, GCOLS) f32
{
  __shared__ __align__(16) _Float16 hpan[2][HPANEL];     // 2 x 64 KB
  __shared__ __align__(16) _Float16 xlds[2][XPANEL];     // 2 x 40 KB
  const int d = blockIdx.x;                              // direction 0..5
  const float* biasTab[6] = {b0, b1, b2, b3, b4, b5};
  const int    colTab[6]  = {1024, 1536, 0, 512, 2048, 2560};
  const float* bias = biasTab[d];
  const _Float16* Bm = Bmats + (size_t)d * KTOT * NGATE;
  const int lane = threadIdx.x & 31;
  const int jt   = threadIdx.x >> 5;                     // j-tile == wave id
  const bool rev = (d & 1);                              // odd dirs run x[::-1]

  // Hoist the 4 per-gate bias scalars (loop-invariant per lane).
  const int nl = lane & 15;
  const float bn_i = bias[0 * KH + jt * 16 + nl];
  const float bn_f = bias[1 * KH + jt * 16 + nl];
  const float bn_g = bias[2 * KH + jt * 16 + nl];
  const float bn_o = bias[3 * KH + jt * 16 + nl];

  // Kick off the TDM for step 0 while we zero the h read panel for step 0.
  if (jt == 0) {
    const int t0 = rev ? (S_STEPS - 1) : 0;
    tdm_load_1d(X + (size_t)t0 * XPANEL,
                (unsigned)(uintptr_t)&xlds[0][0], XPANEL);
  }
  for (int i = threadIdx.x; i < HPANEL; i += blockDim.x)
    hpan[0][i] = (_Float16)0.f;

  // Persistent f32 cell state (WMMA C-layout) + one f32 accumulator set.
  v8f c[4], z[4];
  v8f vz = {};
  #pragma unroll
  for (int mt = 0; mt < 4; ++mt) c[mt] = vz;
  h2 si[4][4];                 // sigmoid(i)        (packed f16, transient)
  h2 ct[4][4];                 // sigmoid(i)*tanh(g)(packed f16, transient)

  const int n_i = 0 * KH + jt * 16;
  const int n_f = 1 * KH + jt * 16;
  const int n_g = 2 * KH + jt * 16;
  const int n_o = 3 * KH + jt * 16;

  for (int t = 0; t < S_STEPS; ++t) {
    if (jt == 0) __builtin_amdgcn_s_wait_tensorcnt(0);   // x panel t arrived
    __syncthreads();            // x panel t usable + h writes of t-1 visible
    if (t + 1 < S_STEPS && jt == 0) {                    // prefetch x panel t+1
      const int tn = rev ? (S_STEPS - 2 - t) : (t + 1);
      tdm_load_1d(X + (size_t)tn * XPANEL,
                  (unsigned)(uintptr_t)&xlds[(t + 1) & 1][0], XPANEL);
    }
    const _Float16* xp = &xlds[t & 1][0];
    const _Float16* hr = &hpan[t & 1][0];                // read panel
    _Float16*       hw = &hpan[(t + 1) & 1][0];          // write panel

    gate_gemm(z, n_i, bn_i, xp, hr, Bm, lane);           // i
    #pragma unroll
    for (int mt = 0; mt < 4; ++mt)
      #pragma unroll
      for (int r = 0; r < 8; ++r)
        si[mt][r >> 1][r & 1] = (_Float16)sigm(z[mt][r]);

    gate_gemm(z, n_g, bn_g, xp, hr, Bm, lane);           // g
    #pragma unroll
    for (int mt = 0; mt < 4; ++mt)
      #pragma unroll
      for (int r = 0; r < 8; ++r)
        ct[mt][r >> 1][r & 1] =
            (_Float16)((float)si[mt][r >> 1][r & 1] * tanhf(z[mt][r]));

    gate_gemm(z, n_f, bn_f, xp, hr, Bm, lane);           // f
    #pragma unroll
    for (int mt = 0; mt < 4; ++mt)
      #pragma unroll
      for (int r = 0; r < 8; ++r)
        c[mt][r] = sigm(z[mt][r]) * c[mt][r] + (float)ct[mt][r >> 1][r & 1];

    gate_gemm(z, n_o, bn_o, xp, hr, Bm, lane);           // o
    {
      const int hi = lane >> 4;
      #pragma unroll
      for (int mt = 0; mt < 4; ++mt)
        #pragma unroll
        for (int r = 0; r < 8; ++r) {
          const int b = mt * 16 + r + 8 * hi;            // C-layout: M=r+8*hi
          hw[b * KH + jt * 16 + nl] =
              (_Float16)(sigm(z[mt][r]) * tanhf(c[mt][r]));
        }
    }
  }

  // Final hidden state lives in hpan[S_STEPS & 1] == hpan[0] (S even).
  __syncthreads();
  {
    const int col0 = colTab[d];
    for (int i = threadIdx.x; i < HPANEL; i += blockDim.x) {
      const int b = i >> 9;                              // / KH
      const int j = i & (KH - 1);
      g_inp[(size_t)b * GCOLS + col0 + j] = (float)hpan[0][i];
    }
  }
}

// ---- Kernel 4: Gumbel gating head (tiny: ~1.5 MFLOP) ----------------------
__global__ void k_head(const float* __restrict__ g_inp,
                       const float* __restrict__ gW, const float* __restrict__ gb,
                       const float* __restrict__ hW, const float* __restrict__ hb,
                       const float* __restrict__ gu, float* __restrict__ out) {
  const int b = threadIdx.x;
  if (b >= B_ROWS) return;
  const float eps = 1e-20f;
  const float* row = g_inp + (size_t)b * GCOLS;

  float lg[3];
  for (int j = 0; j < 3; ++j) {
    const float* w = gW + (size_t)j * GCOLS;
    float s = gb[j];
    for (int k = 0; k < GCOLS; ++k) s = fmaf(row[k], w[k], s);
    lg[j] = fmaxf(s, 0.f);                               // relu
  }
  float mx = fmaxf(lg[0], fmaxf(lg[1], lg[2]));
  float se = 0.f;
  for (int j = 0; j < 3; ++j) se += __expf(lg[j] - mx);
  const float lse = mx + __logf(se);

  float tv[3];
  for (int j = 0; j < 3; ++j) {
    const float u = gu[b * 3 + j];
    const float gum = -__logf(-__logf(u + eps) + eps);
    tv[j] = (lg[j] - lse) + gum;                         // logp + gumbel
  }
  int am = 0;
  if (tv[1] > tv[am]) am = 1;
  if (tv[2] > tv[am]) am = 2;
  float sel[3] = {0.f, 0.f, 0.f};
  sel[am] = 1.f;                                         // selector == y_hard
  const float r1 = sel[0], r2 = sel[1];

  float o[5];
  for (int l = 0; l < 5; ++l) {
    const float* w = hW + (size_t)l * GCOLS;
    float s = hb[l];
    for (int k = 0; k < 1024; ++k)       s = fmaf(row[k] * r1, w[k], s);
    for (int k = 1024; k < GCOLS; ++k)   s = fmaf(row[k] * r2, w[k], s);
    o[l] = s;
  }
  float mx2 = o[0];
  for (int l = 1; l < 5; ++l) mx2 = fmaxf(mx2, o[l]);
  float se2 = 0.f;
  for (int l = 0; l < 5; ++l) se2 += __expf(o[l] - mx2);
  const float lse2 = mx2 + __logf(se2);

  for (int l = 0; l < 5; ++l) out[(size_t)b * 5 + l] = o[l] - lse2;
  for (int j = 0; j < 3; ++j) out[320 + (size_t)b * 3 + j] = sel[j];
}

// ---------------------------------------------------------------------------
extern "C" void kernel_launch(void* const* d_in, const int* in_sizes, int n_in,
                              void* d_out, int out_size, void* d_ws, size_t ws_size,
                              hipStream_t stream) {
  (void)in_sizes; (void)n_in; (void)out_size; (void)ws_size;
  const int*   sent = (const int*)d_in[0];
  const float* emb  = (const float*)d_in[1];
  // directions: 0 inf_f, 1 inf_b, 2 dup_f, 3 dup_b, 4 news_f, 5 news_b
  const float* Wih[6] = {(const float*)d_in[2],  (const float*)d_in[5],
                         (const float*)d_in[8],  (const float*)d_in[11],
                         (const float*)d_in[14], (const float*)d_in[17]};
  const float* Whh[6] = {(const float*)d_in[3],  (const float*)d_in[6],
                         (const float*)d_in[9],  (const float*)d_in[12],
                         (const float*)d_in[15], (const float*)d_in[18]};
  const float* bia[6] = {(const float*)d_in[4],  (const float*)d_in[7],
                         (const float*)d_in[10], (const float*)d_in[13],
                         (const float*)d_in[16], (const float*)d_in[19]};
  const float* gW = (const float*)d_in[20];
  const float* gb = (const float*)d_in[21];
  const float* hW = (const float*)d_in[22];
  const float* hb = (const float*)d_in[23];
  const float* gu = (const float*)d_in[24];

  // workspace layout (~25.3 MB total)
  char* ws = (char*)d_ws;
  const size_t X_BYTES  = (size_t)S_STEPS * B_ROWS * KX * 2;      //  5,242,880
  const size_t BM_BYTES = (size_t)6 * KTOT * NGATE * 2;           // 20,447,232
  _Float16* X   = (_Float16*)(ws);
  _Float16* Bm  = (_Float16*)(ws + X_BYTES);
  float*    gip = (float*)   (ws + X_BYTES + BM_BYTES);

  k_embed<<<S_STEPS * B_ROWS, KX, 0, stream>>>(sent, emb, X);
  for (int dd = 0; dd < 6; ++dd)
    k_wconv<<<KTOT, 256, 0, stream>>>(Wih[dd], Whh[dd],
                                      Bm + (size_t)dd * KTOT * NGATE);
  k_bilstm<<<6, 1024, 0, stream>>>(X, Bm, bia[0], bia[1], bia[2],
                                   bia[3], bia[4], bia[5], gip);
  k_head<<<1, B_ROWS, 0, stream>>>(gip, gW, gb, hW, hb, gu, (float*)d_out);
}
